// Nequip_17815524344049
// MI455X (gfx1250) — matrix operand order repaired and checked by reading, not compiled, not run on hardware
//
#include <hip/hip_runtime.h>
#include <stdint.h>
#include <stddef.h>

#define NN 20000
#define EE 640000
#define AD 64
#define FD 32
#define BD 8
#define HD 64
#define NP 5

#define INV_SQRT_AVG 0.17677669529663687f /* 1/sqrt(32) */

typedef __attribute__((ext_vector_type(16))) __bf16 v16bf;
typedef __attribute__((ext_vector_type(8)))  float  v8f;
typedef __attribute__((ext_vector_type(4)))  uint32_t v4u;
typedef __attribute__((ext_vector_type(8)))  int      v8i_t;
typedef __attribute__((ext_vector_type(4)))  int      v4i_t;

union Frag {
    uint32_t u[8];
    uint4    q[2];
    v16bf    h;
};

/* exact RNE f32->bf16 pair (used in one-time weight packing) */
__device__ __forceinline__ uint32_t pack2bf(float lo, float hi) {
    uint32_t a = __float_as_uint(lo);
    uint32_t b = __float_as_uint(hi);
    uint32_t ra = (a + 0x7FFFu + ((a >> 16) & 1u)) >> 16;
    uint32_t rb = (b + 0x7FFFu + ((b >> 16) & 1u)) >> 16;
    return (ra & 0xFFFFu) | (rb << 16);
}

/* fast pack: round-half-up + single v_perm_b32 byte-select */
__device__ __forceinline__ uint32_t pack2bf_fast(float lo, float hi) {
    return __builtin_amdgcn_perm(__float_as_uint(hi) + 0x8000u,
                                 __float_as_uint(lo) + 0x8000u,
                                 0x07060302u);
}

__device__ __forceinline__ v8f wmma_bf16(const Frag& a, const Frag& b, v8f c) {
    return __builtin_amdgcn_wmma_f32_16x16x32_bf16(false, a.h, false, b.h,
                                                   (short)0, c, false, false);
}

// B fragment: dst[((ks*NT+t)*32+lane)*8 + i] holds bf16 pair
//   (B[ks*32 + (lane>>4)*16 + 2i,  t*16 + (lane&15)], B[k+1, same col])
__device__ __forceinline__ Frag load_bfrag(const uint32_t* __restrict__ base,
                                           int ks, int NT, int t, int lane) {
    Frag b;
    const uint32_t* p = base + (((size_t)(ks * NT + t) * 32 + lane) << 3);
    b.q[0] = *(const uint4*)p;
    b.q[1] = *(const uint4*)(p + 4);
    return b;
}

__device__ __forceinline__ Frag load_bfrag_lds(const uint32_t* base,
                                               int ks, int NT, int t, int lane) {
    Frag b;
    const uint32_t* p = base + (((ks * NT + t) * 32 + lane) << 3);
    b.q[0] = *(const uint4*)p;
    b.q[1] = *(const uint4*)(p + 4);
    return b;
}

// A-fragment k order used here (contiguous, same as B packing): lane half h = lane>>4, row m = lane&15.
// VGPR i: K = 16h + 2i  (pairs K,K+1)
__device__ __forceinline__ int a_relk(int i, int half) {
    return 16 * half + 2 * i;
}

__device__ __forceinline__ float siluf(float x)    { return x / (1.0f + __expf(-x)); }
__device__ __forceinline__ float sigmoidf_(float x){ return 1.0f / (1.0f + __expf(-x)); }

/* ---------------- weight pack: f32 row-major (Ksrc x Ncols) -> WMMA B frags -------- */
__global__ void pack_b_kernel(const float* __restrict__ src, int Ksrc, int Ncols,
                              int KS, int NT, uint32_t* __restrict__ dst) {
    int idx = blockIdx.x * 256 + threadIdx.x;
    int total = KS * NT * 256;
    if (idx >= total) return;
    int i    = idx & 7;
    int lane = (idx >> 3) & 31;
    int rest = idx >> 8;
    int t    = rest % NT;
    int ks   = rest / NT;
    int col  = t * 16 + (lane & 15);
    int k    = ks * 32 + (lane >> 4) * 16 + 2 * i;
    float lo = (k     < Ksrc) ? src[(size_t)k * Ncols + col]       : 0.0f;
    float hi = (k + 1 < Ksrc) ? src[(size_t)(k + 1) * Ncols + col] : 0.0f;
    dst[idx] = pack2bf(lo, hi);
}

/* ---------------- edge geometry: bessel*fcut (bf16 rows) + Y1 --------------------- */
__global__ void geom_kernel(const float* __restrict__ coords, const int* __restrict__ ei,
                            uint32_t* __restrict__ efeat, float* __restrict__ y1) {
    int e = blockIdx.x * 256 + threadIdx.x;
    if (e >= EE) return;
    int sn = ei[e], rv = ei[EE + e];
    float vx = coords[3 * sn + 0] - coords[3 * rv + 0];
    float vy = coords[3 * sn + 1] - coords[3 * rv + 1];
    float vz = coords[3 * sn + 2] - coords[3 * rv + 2];
    float r = sqrtf(vx * vx + vy * vy + vz * vz + 1e-12f);
    float rinv = 1.0f / r;
    float x = r * 0.2f;                       // r / RMAX
    float x2 = x * x, x3 = x2 * x, x6 = x3 * x3;
    float fc = (x < 1.0f) ? (1.0f - 28.0f * x6 + 48.0f * x6 * x - 21.0f * x6 * x2) : 0.0f;
    float pref = 0.6324555320336759f * rinv * fc;   // sqrt(2/RMAX)
    const float pi = 3.14159265358979323846f;
    float f[8];
#pragma unroll
    for (int n = 0; n < 8; n++) f[n] = pref * __sinf((float)(n + 1) * pi * x);
    uint4 pk;
    pk.x = pack2bf(f[0], f[1]); pk.y = pack2bf(f[2], f[3]);
    pk.z = pack2bf(f[4], f[5]); pk.w = pack2bf(f[6], f[7]);
    *(uint4*)(efeat + (size_t)e * 4) = pk;
    float s3 = 1.7320508075688772f * rinv;
    y1[3 * e + 0] = s3 * vx; y1[3 * e + 1] = s3 * vy; y1[3 * e + 2] = s3 * vz;
}

/* ---------------- node embedding: s = attrs(Nx64) @ W(64x32), WMMA ---------------- */
__global__ __launch_bounds__(64) void embed_kernel(const float* __restrict__ attrs,
                                                   const uint32_t* __restrict__ embB,
                                                   float* __restrict__ s) {
    int wave = threadIdx.x >> 5, lane = threadIdx.x & 31;
    int nb = (blockIdx.x * 2 + wave) * 16;
    int m = lane & 15, half = lane >> 4;
    int anode = nb + m;
    v8f acc0 = {}, acc1 = {};
#pragma unroll
    for (int ks = 0; ks < 2; ks++) {
        Frag a;
#pragma unroll
        for (int i = 0; i < 8; i++) {
            int k = ks * 32 + a_relk(i, half);
            float2 p = *(const float2*)(attrs + (size_t)anode * AD + k);
            a.u[i] = pack2bf_fast(p.x, p.y);
        }
        Frag b0 = load_bfrag(embB, ks, 2, 0, lane);
        Frag b1 = load_bfrag(embB, ks, 2, 1, lane);
        acc0 = wmma_bf16(a, b0, acc0);
        acc1 = wmma_bf16(a, b1, acc1);
    }
    int col = lane & 15;
#pragma unroll
    for (int r = 0; r < 8; r++) {
        int row = nb + half * 8 + r;
        s[(size_t)row * FD + col]      = acc0[r];
        s[(size_t)row * FD + col + 16] = acc1[r];
    }
}

/* ---------------- fused edge kernel: radial MLP (WMMA) + messages + scatter -------
 * GEMM2 weights (packed rad_W2, 20 KB, shared by all waves) are staged into LDS
 * once per workgroup with the Tensor Data Mover (TDM). */
#define MW 4            /* waves (16-edge tiles) per block */
#define W2DW (2*10*256) /* 5120 dwords of packed W2 fragments */

__global__ __launch_bounds__(128) void message_kernel(
        const int* __restrict__ ei, const uint32_t* __restrict__ efeat,
        const float* __restrict__ y1,
        const float* __restrict__ s, const float* __restrict__ v,
        const uint32_t* __restrict__ W1B, const float* __restrict__ b1,
        const uint32_t* __restrict__ W2B,
        float* __restrict__ aggS, float* __restrict__ aggV) {
    // h (16x64) for GEMM2 A-build is aliased into the first 64 columns of wbuf;
    // the final w (16x160) overwrites it only after the A fragments are consumed
    // (same-wave DS ordering).
    __shared__ float wbuf[MW][16][NP * FD];                 // 40 KB
    __shared__ __align__(16) uint32_t w2lds[W2DW];          // 20 KB TDM-staged W2
    __shared__ int   esnd[MW][16];
    __shared__ int   ercv[MW][16];
    __shared__ float ey[MW][16][3];

    int wave = threadIdx.x >> 5, lane = threadIdx.x & 31;
    int ebase = (blockIdx.x * MW + wave) * 16;
    int m = lane & 15, half = lane >> 4, col = lane & 15;

    if (lane < 16) {
        int e = ebase + lane;
        esnd[wave][lane] = ei[e];
        ercv[wave][lane] = ei[EE + e];
        ey[wave][lane][0] = y1[3 * e + 0];
        ey[wave][lane][1] = y1[3 * e + 1];
        ey[wave][lane][2] = y1[3 * e + 2];
    }

    // ---- TDM: wave0 DMAs the packed W2 block (5120 dwords) into LDS -------------
    if (threadIdx.x < 32) {
        uint64_t ga = (uint64_t)(uintptr_t)W2B;
        uint32_t ldsa = (uint32_t)(uintptr_t)&w2lds[0];
        v4u g0;
        g0[0] = 1u;                                          // count=1 (valid user D#)
        g0[1] = ldsa;                                        // lds_addr (bytes)
        g0[2] = (uint32_t)ga;                                // global_addr[31:0]
        g0[3] = (uint32_t)((ga >> 32) & 0x01FFFFFFu) | (2u << 30); // ga[56:32] | type=2
        v8i_t g1;
        g1[0] = (int)(2u << 16);                             // data_size = 4 bytes
        g1[1] = (int)((W2DW & 0xFFFFu) << 16);               // tensor_dim0 lo16
        g1[2] = (int)((W2DW >> 16) | (1u << 16));            // tensor_dim0 hi | dim1=1
        g1[3] = (int)((uint32_t)W2DW << 16);                 // tile_dim0 = 5120
        g1[4] = 1;                                           // tile_dim1 = 1
        g1[5] = W2DW;                                        // tensor_dim0_stride
        g1[6] = 0;
        g1[7] = 0;
        v4i_t gz = {0, 0, 0, 0};
#if defined(__clang_major__) && (__clang_major__ >= 23)
        v8i_t gz8 = {0, 0, 0, 0, 0, 0, 0, 0};
        __builtin_amdgcn_tensor_load_to_lds(g0, g1, gz, gz, gz8, 0);
#else
        __builtin_amdgcn_tensor_load_to_lds(g0, g1, gz, gz, 0);
#endif
        __builtin_amdgcn_s_wait_tensorcnt(0);
    }
    __syncthreads();

    // ---- GEMM1: edge_feats (16x8, K padded to 32) @ W1 (8x64) -> h, bias+SiLU ----
    Frag a1;
    uint4 fe = make_uint4(0u, 0u, 0u, 0u);
    if (half == 0) fe = *(const uint4*)(efeat + (size_t)(ebase + m) * 4);
    a1.q[0] = fe;
    a1.q[1] = make_uint4(0u, 0u, 0u, 0u);
    v8f h4[4] = {{}, {}, {}, {}};
#pragma unroll
    for (int t = 0; t < 4; t++) {
        Frag b = load_bfrag(W1B, 0, 4, t, lane);
        h4[t] = wmma_bf16(a1, b, h4[t]);
    }
#pragma unroll
    for (int t = 0; t < 4; t++) {
        float bias = b1[t * 16 + col];
#pragma unroll
        for (int r = 0; r < 8; r++) {
            float hv = h4[t][r] + bias;
            wbuf[wave][half * 8 + r][t * 16 + col] = siluf(hv);   // h region (cols 0..63)
        }
    }

    // ---- GEMM2: h (16x64) @ W2 (64x160) -> w (16x160) ----------------------------
    v8f w10[10] = {{}, {}, {}, {}, {}, {}, {}, {}, {}, {}};
#pragma unroll
    for (int ks = 0; ks < 2; ks++) {
        Frag a;
#pragma unroll
        for (int i = 0; i < 8; i++) {
            int k = ks * 32 + a_relk(i, half);
            float2 p = *(const float2*)&wbuf[wave][m][k];          // read h
            a.u[i] = pack2bf_fast(p.x, p.y);
        }
#pragma unroll
        for (int t = 0; t < 10; t++) {
            Frag b = load_bfrag_lds(w2lds, ks, 10, t, lane);
            w10[t] = wmma_bf16(a, b, w10[t]);
        }
    }
#pragma unroll
    for (int t = 0; t < 10; t++)
#pragma unroll
        for (int r = 0; r < 8; r++)
            wbuf[wave][half * 8 + r][t * 16 + col] = w10[t][r];

    // ---- scalar phase: lane = feature f, loop edges; messages + atomic scatter ---
    int f = lane;
#pragma unroll 1
    for (int e = 0; e < 16; e++) {
        float w0 = wbuf[wave][e][0 * FD + f];
        float w1 = wbuf[wave][e][1 * FD + f];
        float w2 = wbuf[wave][e][2 * FD + f];
        float w3 = wbuf[wave][e][3 * FD + f];
        float w4 = wbuf[wave][e][4 * FD + f];
        int sn = esnd[wave][e];
        int rv = ercv[wave][e];
        float yx = ey[wave][e][0], yy = ey[wave][e][1], yz = ey[wave][e][2];
        float sj = s[(size_t)sn * FD + f];
        const float* vp = v + ((size_t)sn * FD + f) * 3;
        float vjx = vp[0], vjy = vp[1], vjz = vp[2];
        float dot = vjx * yx + vjy * yy + vjz * yz;
        float cx = vjy * yz - vjz * yy;
        float cy = vjz * yx - vjx * yz;
        float cz = vjx * yy - vjy * yx;
        float ms  = (w0 * sj + w3 * dot) * INV_SQRT_AVG;
        float sjw = w1 * sj;
        float mvx = (sjw * yx + w2 * vjx + w4 * cx) * INV_SQRT_AVG;
        float mvy = (sjw * yy + w2 * vjy + w4 * cy) * INV_SQRT_AVG;
        float mvz = (sjw * yz + w2 * vjz + w4 * cz) * INV_SQRT_AVG;
        atomicAdd(&aggS[(size_t)rv * FD + f], ms);
        atomicAdd(&aggV[0 * (size_t)NN * FD + (size_t)rv * FD + f], mvx);
        atomicAdd(&aggV[1 * (size_t)NN * FD + (size_t)rv * FD + f], mvy);
        atomicAdd(&aggV[2 * (size_t)NN * FD + (size_t)rv * FD + f], mvz);
    }
}

/* ---------------- node update: sc einsums (K=2048 WMMA) + lin + gated residual ---- */
__global__ __launch_bounds__(64) void node_kernel(
        const float* __restrict__ attrs,
        float* __restrict__ s, float* __restrict__ v,
        const float* __restrict__ aggS, const float* __restrict__ aggV,
        const uint32_t* __restrict__ scWsB, const uint32_t* __restrict__ scWvB,
        const uint32_t* __restrict__ linWsB, const uint32_t* __restrict__ linWvB) {
    int wave = threadIdx.x >> 5, lane = threadIdx.x & 31;
    int nb = (blockIdx.x * 2 + wave) * 16;
    int m = lane & 15, half = lane >> 4, col = lane & 15;
    int anode = nb + m;

    // -------- pre_s = outer(s,attrs) @ sc_Ws  +  agg_s @ lin_Ws   (16 x 64)
    v8f acc[4] = {{}, {}, {}, {}};
#pragma unroll 2
    for (int ks = 0; ks < 64; ks++) {
        int fidx  = ks >> 1;
        int abase = (ks & 1) * 32;
        float sval = s[(size_t)anode * FD + fidx];
        Frag a;
#pragma unroll
        for (int i = 0; i < 8; i++) {
            int rk = a_relk(i, half);
            float2 at = *(const float2*)(attrs + (size_t)anode * AD + abase + rk);
            a.u[i] = pack2bf_fast(sval * at.x, sval * at.y);
        }
#pragma unroll
        for (int t = 0; t < 4; t++) {
            Frag b = load_bfrag(scWsB, ks, 4, t, lane);
            acc[t] = wmma_bf16(a, b, acc[t]);
        }
    }
    {   // lin_Ws k-step (K=32)
        Frag a;
#pragma unroll
        for (int i = 0; i < 8; i++) {
            int rk = a_relk(i, half);
            float2 ag = *(const float2*)(aggS + (size_t)anode * FD + rk);
            a.u[i] = pack2bf_fast(ag.x, ag.y);
        }
#pragma unroll
        for (int t = 0; t < 4; t++) {
            Frag b = load_bfrag(linWsB, 0, 4, t, lane);
            acc[t] = wmma_bf16(a, b, acc[t]);
        }
    }
    // epilogue: s update + keep sigmoid(gates) in registers
    float sig0[8], sig1[8];
#pragma unroll
    for (int r = 0; r < 8; r++) {
        int row = nb + half * 8 + r;
        float f0 = acc[0][r], f1 = acc[1][r];
        float g0 = acc[2][r], g1 = acc[3][r];
        float* sp0 = &s[(size_t)row * FD + col];
        float* sp1 = &s[(size_t)row * FD + col + 16];
        *sp0 = siluf(f0) + *sp0;
        *sp1 = siluf(f1) + *sp1;
        sig0[r] = sigmoidf_(g0);
        sig1[r] = sigmoidf_(g1);
    }

    // -------- pre_v per component: outer(v_c,attrs) @ sc_Wv + agg_v_c @ lin_Wv (16x32)
#pragma unroll 1
    for (int c = 0; c < 3; c++) {
        v8f av0 = {}, av1 = {};
#pragma unroll 2
        for (int ks = 0; ks < 64; ks++) {
            int fidx  = ks >> 1;
            int abase = (ks & 1) * 32;
            float vval = v[((size_t)anode * FD + fidx) * 3 + c];
            Frag a;
#pragma unroll
            for (int i = 0; i < 8; i++) {
                int rk = a_relk(i, half);
                float2 at = *(const float2*)(attrs + (size_t)anode * AD + abase + rk);
                a.u[i] = pack2bf_fast(vval * at.x, vval * at.y);
            }
            Frag b0 = load_bfrag(scWvB, ks, 2, 0, lane);
            Frag b1 = load_bfrag(scWvB, ks, 2, 1, lane);
            av0 = wmma_bf16(a, b0, av0);
            av1 = wmma_bf16(a, b1, av1);
        }
        {   // lin_Wv k-step
            Frag a;
#pragma unroll
            for (int i = 0; i < 8; i++) {
                int rk = a_relk(i, half);
                float2 ag = *(const float2*)(aggV + (size_t)c * NN * FD +
                                             (size_t)anode * FD + rk);
                a.u[i] = pack2bf_fast(ag.x, ag.y);
            }
            Frag b0 = load_bfrag(linWvB, 0, 2, 0, lane);
            Frag b1 = load_bfrag(linWvB, 0, 2, 1, lane);
            av0 = wmma_bf16(a, b0, av0);
            av1 = wmma_bf16(a, b1, av1);
        }
#pragma unroll
        for (int r = 0; r < 8; r++) {
            int row = nb + half * 8 + r;
            float* vp0 = &v[((size_t)row * FD + col) * 3 + c];
            float* vp1 = &v[((size_t)row * FD + col + 16) * 3 + c];
            *vp0 = av0[r] * sig0[r] + *vp0;
            *vp1 = av1[r] * sig1[r] + *vp1;
        }
    }
}

/* ---------------- output: concat [s, v.reshape(N,96)] ----------------------------- */
__global__ void out_kernel(const float* __restrict__ s, const float* __restrict__ v,
                           float* __restrict__ out) {
    int i = blockIdx.x * 256 + threadIdx.x;
    if (i >= NN * 128) return;
    int n = i >> 7, j = i & 127;
    out[i] = (j < FD) ? s[(size_t)n * FD + j] : v[(size_t)n * FD * 3 + (j - FD)];
}

/* ================================ launch ========================================== */
extern "C" void kernel_launch(void* const* d_in, const int* in_sizes, int n_in,
                              void* d_out, int out_size, void* d_ws, size_t ws_size,
                              hipStream_t stream) {
    (void)in_sizes; (void)n_in; (void)out_size; (void)ws_size;
    const float* attrs  = (const float*)d_in[0];
    const float* coords = (const float*)d_in[1];
    const int*   ei     = (const int*)d_in[2];
    const float* embW   = (const float*)d_in[3];
    const float* radW1  = (const float*)d_in[4];
    const float* radb1  = (const float*)d_in[5];
    const float* radW2  = (const float*)d_in[6];
    const float* linWs  = (const float*)d_in[7];
    const float* linWv  = (const float*)d_in[8];
    const float* scWs   = (const float*)d_in[9];
    const float* scWv   = (const float*)d_in[10];
    float* out = (float*)d_out;

    char* ws = (char*)d_ws;
    size_t off = 0;
    auto alloc = [&](size_t bytes) -> void* {
        void* p = ws + off;
        off += bytes;
        off = (off + 255) & ~(size_t)255;
        return p;
    };
    float*    sbuf  = (float*)alloc((size_t)NN * FD * 4);
    float*    vbuf  = (float*)alloc((size_t)NN * FD * 3 * 4);
    float*    aggS  = (float*)alloc((size_t)NN * FD * 4);
    float*    aggV  = (float*)alloc((size_t)3 * NN * FD * 4);
    float*    y1    = (float*)alloc((size_t)EE * 3 * 4);
    uint32_t* efeat = (uint32_t*)alloc((size_t)EE * 4 * 4);
    uint32_t* embB  = (uint32_t*)alloc((size_t)2 * 2 * 256 * 4);
    uint32_t *W1B[2], *W2B[2], *lsB[2], *lvB[2], *ssB[2], *svB[2];
    for (int l = 0; l < 2; l++) {
        W1B[l] = (uint32_t*)alloc((size_t)1 * 4 * 256 * 4);
        W2B[l] = (uint32_t*)alloc((size_t)2 * 10 * 256 * 4);
        lsB[l] = (uint32_t*)alloc((size_t)1 * 4 * 256 * 4);
        lvB[l] = (uint32_t*)alloc((size_t)1 * 2 * 256 * 4);
        ssB[l] = (uint32_t*)alloc((size_t)64 * 4 * 256 * 4);
        svB[l] = (uint32_t*)alloc((size_t)64 * 2 * 256 * 4);
    }

    auto pack = [&](const float* src, int Ks, int Nc, int KS, int NT, uint32_t* dst) {
        int total = KS * NT * 256;
        pack_b_kernel<<<(total + 255) / 256, 256, 0, stream>>>(src, Ks, Nc, KS, NT, dst);
    };
    pack(embW, 64, 32, 2, 2, embB);
    for (int l = 0; l < 2; l++) {
        pack(radW1 + (size_t)l * BD * HD,           8,   64,  1,  4, W1B[l]);
        pack(radW2 + (size_t)l * HD * NP * FD,      64,  160, 2, 10, W2B[l]);
        pack(linWs + (size_t)l * FD * 2 * FD,       32,  64,  1,  4, lsB[l]);
        pack(linWv + (size_t)l * FD * FD,           32,  32,  1,  2, lvB[l]);
        pack(scWs  + (size_t)l * FD * AD * 2 * FD,  2048, 64, 64, 4, ssB[l]);
        pack(scWv  + (size_t)l * FD * AD * FD,      2048, 32, 64, 2, svB[l]);
    }

    hipMemsetAsync(vbuf, 0, (size_t)NN * FD * 3 * 4, stream);
    geom_kernel<<<(EE + 255) / 256, 256, 0, stream>>>(coords, ei, efeat, y1);
    embed_kernel<<<NN / 32, 64, 0, stream>>>(attrs, embB, sbuf);

    for (int l = 0; l < 2; l++) {
        hipMemsetAsync(aggS, 0, (size_t)NN * FD * 4, stream);
        hipMemsetAsync(aggV, 0, (size_t)3 * NN * FD * 4, stream);
        message_kernel<<<EE / (16 * MW), 128, 0, stream>>>(
            ei, efeat, y1, sbuf, vbuf,
            W1B[l], radb1 + (size_t)l * HD, W2B[l], aggS, aggV);
        node_kernel<<<NN / 32, 64, 0, stream>>>(
            attrs, sbuf, vbuf, aggS, aggV, ssB[l], svB[l], lsB[l], lvB[l]);
    }

    out_kernel<<<(NN * 128 + 255) / 256, 256, 0, stream>>>(sbuf, vbuf, out);
}
